// SparseAttention1_12919261626595
// MI455X (gfx1250) — matrix operand order repaired and checked by reading, not compiled, not run on hardware
//
#include <hip/hip_runtime.h>
#include <cstdint>
#include <cstddef>

typedef __attribute__((ext_vector_type(16))) _Float16 v16h;
typedef __attribute__((ext_vector_type(8)))  float    v8f;

#define B_    8
#define H_    12
#define S_    1024
#define D_    64
#define BN    64           // keys per inner iteration (2 WMMA K-chunks)
#define WAVES 8
#define BM    (WAVES * 16) // 128 query rows per block
#define NCH   (S_ / BN)    // 16 key chunks

union AF {
    v16h     v;
    uint32_t u[8];
    _Float16 e[16];
};

__device__ __forceinline__ v8f wmma16x16x32(v16h a, v16h b, v8f c) {
    // D = A(16x32 f16) * B(32x16 f16) + C(16x16 f32)
    return __builtin_amdgcn_wmma_f32_16x16x32_f16(
        /*neg_a=*/false, a, /*neg_b=*/false, b,
        /*c_mod=*/(short)0, c, /*reuse_a=*/false, /*reuse_b=*/false);
}

__global__ __launch_bounds__(256)
void fa_masked_kernel(const float* __restrict__ Q,
                      const float* __restrict__ K,
                      const float* __restrict__ V,
                      const int*   __restrict__ mask,
                      float*       __restrict__ Out)
{
    // Ping-pong staging buffers (software pipeline over key chunks)
    __shared__ _Float16 Kb[2][BN * D_];          // [key][d] row-major  (2 x 8 KB)
    __shared__ _Float16 Vb[2][D_ * BN];          // [d][key] transposed (2 x 8 KB)
    __shared__ float    biasLds[2][BN];          // -1e6*(1-mask)
    __shared__ _Float16 Pb[WAVES * 16 * BN];     // per-wave P staging  (16 KB)

    const int tid  = threadIdx.x;
    const int lane = tid & 31;
    const int w    = tid >> 5;
    const int row  = lane & 15;   // N index in 16-wide tile / A-matrix M
    const int hi   = lane >> 4;   // lane half selector

    const int bm = blockIdx.x % (S_ / BM);
    const int bh = blockIdx.x / (S_ / BM);       // fused (b,h), 0..95
    const int b  = bh / H_;

    const int q0 = bm * BM + w * 16;             // this wave's first query row

    const float* Kbase = K + (size_t)bh * S_ * D_;
    const float* Vbase = V + (size_t)bh * S_ * D_;
    const int*   mbase = mask + (size_t)b * S_;
    _Float16*    Pw    = Pb + w * 16 * BN;

    // ---- Load Q fragments (A-matrix layout), folding in scale = 1/sqrt(64) ----
    AF qa[2];
    {
        const float scale = 0.125f;
        const float* qbase = Q + ((size_t)bh * S_ + (size_t)(q0 + row)) * D_;
        #pragma unroll
        for (int c = 0; c < 2; ++c) {
            #pragma unroll
            for (int v = 0; v < 8; ++v) {
                const int kk = (v < 4) ? (8 * hi + 2 * v) : (16 + 8 * hi + 2 * (v - 4));
                const int d  = c * 32 + kk;
                qa[c].e[2 * v]     = (_Float16)(qbase[d]     * scale);
                qa[c].e[2 * v + 1] = (_Float16)(qbase[d + 1] * scale);
            }
        }
    }

    // All-ones B fragment: rowsum(P) = P x ones
    AF ones;
    #pragma unroll
    for (int i = 0; i < 16; ++i) ones.e[i] = (_Float16)1.0f;

    v8f o0 = {}, o1 = {}, o2 = {}, o3 = {};      // O accumulators (16 x 64)
    float mrow[8], lrow[8];
    #pragma unroll
    for (int j = 0; j < 8; ++j) { mrow[j] = -__builtin_inff(); lrow[j] = 0.0f; }

    // ---- Pipeline prologue: fetch chunk 0 into registers, stage to buffer 0 ----
    float2 kreg[8], vreg[8];
    float  breg = 0.0f;
    {
        const float2* K2 = (const float2*)Kbase;
        const float2* V2 = (const float2*)Vbase;
        #pragma unroll
        for (int r = 0; r < 8; ++r) {
            kreg[r] = K2[tid + r * 256];
            vreg[r] = V2[tid + r * 256];
        }
        if (tid < BN) breg = -1.0e6f * (1.0f - (float)mbase[tid]);
    }
    #pragma unroll
    for (int r = 0; r < 8; ++r) {
        const int i2  = tid + r * 256;
        const int key = i2 >> 5;
        const int d   = (2 * i2) & 63;
        Kb[0][2 * i2]     = (_Float16)kreg[r].x;
        Kb[0][2 * i2 + 1] = (_Float16)kreg[r].y;
        Vb[0][d * BN + key]       = (_Float16)vreg[r].x;
        Vb[0][(d + 1) * BN + key] = (_Float16)vreg[r].y;
    }
    if (tid < BN) biasLds[0][tid] = breg;
    __syncthreads();

    for (int kb = 0; kb < NCH; ++kb) {
        const int  cur  = kb & 1;
        const bool more = (kb + 1) < NCH;

        // ---- Issue next chunk's global loads early (latency hidden by compute) ----
        if (more) {
            const size_t nchunk = (size_t)(kb + 1) * BN * D_;
            const float2* K2 = (const float2*)(Kbase + nchunk);
            const float2* V2 = (const float2*)(Vbase + nchunk);
            #pragma unroll
            for (int r = 0; r < 8; ++r) {
                kreg[r] = K2[tid + r * 256];
                vreg[r] = V2[tid + r * 256];
            }
            if (tid < BN) breg = -1.0e6f * (1.0f - (float)mbase[(kb + 1) * BN + tid]);
        }

        // ---- Scores: 4 key tiles x 2 K-chunks -> 8 WMMAs ----
        v8f s[4];
        #pragma unroll
        for (int nt = 0; nt < 4; ++nt) s[nt] = (v8f){};
        {
            const uint32_t* Kb32 = (const uint32_t*)Kb[cur];
            #pragma unroll
            for (int c = 0; c < 2; ++c) {
                #pragma unroll
                for (int nt = 0; nt < 4; ++nt) {
                    AF kf;
                    #pragma unroll
                    for (int v = 0; v < 8; ++v)
                        kf.u[v] = Kb32[(nt * 16 + row) * 32 + c * 16 + ((v < 4) ? (4 * hi + v) : (8 + 4 * hi + (v - 4)))];
                    s[nt] = wmma16x16x32(qa[c].v, kf.v, s[nt]);
                }
            }
        }

        // ---- Add mask bias ----
        #pragma unroll
        for (int nt = 0; nt < 4; ++nt) {
            const float bnt = biasLds[cur][nt * 16 + row];
            #pragma unroll
            for (int j = 0; j < 8; ++j) s[nt][j] += bnt;
        }

        // ---- Row max: lane-local over 4 tiles, then batched 16-lane butterfly ----
        float mx[8];
        #pragma unroll
        for (int j = 0; j < 8; ++j)
            mx[j] = fmaxf(fmaxf(s[0][j], s[1][j]), fmaxf(s[2][j], s[3][j]));
        #pragma unroll
        for (int step = 1; step < 16; step <<= 1) {
            float t[8];
            #pragma unroll
            for (int j = 0; j < 8; ++j) t[j] = __shfl_xor(mx[j], step, 32);
            #pragma unroll
            for (int j = 0; j < 8; ++j) mx[j] = fmaxf(mx[j], t[j]);
        }
        float alpha[8];
        #pragma unroll
        for (int j = 0; j < 8; ++j) {
            const float mnew = fmaxf(mx[j], mrow[j]);
            alpha[j] = __expf(mrow[j] - mnew);
            mrow[j]  = mnew;
        }

        // ---- P = exp(s - m), stored into per-wave LDS (C -> A relayout) ----
        #pragma unroll
        for (int j = 0; j < 8; ++j) {
            const int m = j + 8 * hi;
            #pragma unroll
            for (int nt = 0; nt < 4; ++nt)
                Pw[m * BN + nt * 16 + row] = (_Float16)__expf(s[nt][j] - mrow[j]);
        }
        asm volatile("s_wait_dscnt 0" ::: "memory");

        // ---- Rebuild P as A fragments (2 K-chunks of 32) ----
        AF pa[2];
        {
            const uint32_t* Pw32 = (const uint32_t*)Pw;
            #pragma unroll
            for (int kc = 0; kc < 2; ++kc)
                #pragma unroll
                for (int v = 0; v < 8; ++v) {
                    const int kh = (v < 4) ? (4 * hi + v) : (8 + 4 * hi + (v - 4));
                    pa[kc].u[v] = Pw32[row * 32 + kc * 16 + kh];
                }
        }

        // ---- Row sums via WMMA: rs(m, n) = rowsum(m) for every n ----
        v8f rs = {};
        rs = wmma16x16x32(pa[0].v, ones.v, rs);
        rs = wmma16x16x32(pa[1].v, ones.v, rs);

        // ---- Update l, rescale O ----
        #pragma unroll
        for (int j = 0; j < 8; ++j) {
            lrow[j] = lrow[j] * alpha[j] + rs[j];
            o0[j] *= alpha[j]; o1[j] *= alpha[j];
            o2[j] *= alpha[j]; o3[j] *= alpha[j];
        }

        // ---- O += P . V  (4 N-tiles over D=64, 2 K-chunks) -> 8 WMMAs ----
        {
            const uint32_t* Vb32 = (const uint32_t*)Vb[cur];
            #pragma unroll
            for (int kc = 0; kc < 2; ++kc) {
                #pragma unroll
                for (int nt = 0; nt < 4; ++nt) {
                    AF vb;
                    #pragma unroll
                    for (int v = 0; v < 8; ++v)
                        vb.u[v] = Vb32[(nt * 16 + row) * 32 + kc * 16 + ((v < 4) ? (4 * hi + v) : (8 + 4 * hi + (v - 4)))];
                    v8f acc = (nt == 0) ? o0 : (nt == 1) ? o1 : (nt == 2) ? o2 : o3;
                    acc = wmma16x16x32(pa[kc].v, vb.v, acc);
                    if      (nt == 0) o0 = acc;
                    else if (nt == 1) o1 = acc;
                    else if (nt == 2) o2 = acc;
                    else              o3 = acc;
                }
            }
        }

        // ---- Stage next chunk into the other buffer ----
        if (more) {
            const int nxt = 1 - cur;
            #pragma unroll
            for (int r = 0; r < 8; ++r) {
                const int i2  = tid + r * 256;
                const int key = i2 >> 5;
                const int d   = (2 * i2) & 63;
                Kb[nxt][2 * i2]     = (_Float16)kreg[r].x;
                Kb[nxt][2 * i2 + 1] = (_Float16)kreg[r].y;
                Vb[nxt][d * BN + key]       = (_Float16)vreg[r].x;
                Vb[nxt][(d + 1) * BN + key] = (_Float16)vreg[r].y;
            }
            if (tid < BN) biasLds[nxt][tid] = breg;
        }
        // Single barrier: publishes writes(kb+1 buffer) AND separates this
        // iteration's reads of buffer `cur` from its overwrite at kb+2.
        __syncthreads();
    }

    // ---- Normalize by row sums and write out [B,H,S,D] ----
    float rinv[8];
    #pragma unroll
    for (int j = 0; j < 8; ++j) rinv[j] = 1.0f / lrow[j];

    float* obase = Out + ((size_t)bh * S_ + (size_t)q0) * D_;
    #pragma unroll
    for (int j = 0; j < 8; ++j) {
        const int m = j + 8 * hi;
        float* orow = obase + (size_t)m * D_;
        orow[0 * 16 + row] = o0[j] * rinv[j];
        orow[1 * 16 + row] = o1[j] * rinv[j];
        orow[2 * 16 + row] = o2[j] * rinv[j];
        orow[3 * 16 + row] = o3[j] * rinv[j];
    }
}

extern "C" void kernel_launch(void* const* d_in, const int* in_sizes, int n_in,
                              void* d_out, int out_size, void* d_ws, size_t ws_size,
                              hipStream_t stream) {
    (void)in_sizes; (void)n_in; (void)out_size; (void)d_ws; (void)ws_size;
    const float* Q    = (const float*)d_in[0];
    const float* K    = (const float*)d_in[1];
    const float* V    = (const float*)d_in[2];
    // d_in[3] route_mat, d_in[4] ids: grouping is a batch permutation; outputs
    // are returned in original batch order, so they do not affect the math.
    const int*   mask = (const int*)d_in[5];
    float*       out  = (float*)d_out;

    dim3 grid(B_ * H_ * (S_ / BM));   // 768 blocks
    dim3 block(256);                  // 8 wave32 waves
    fa_masked_kernel<<<grid, block, 0, stream>>>(Q, K, V, mask, out);
}
